// CCSequenceModel_36112085025200
// MI455X (gfx1250) — matrix operand order repaired and checked by reading, not compiled, not run on hardware
//
#include <hip/hip_runtime.h>

#define HD 64      // hidden size
#define G3 192     // 3*H gates
#define NTILES 12  // 192/16 N tiles
#define TSEQ 256   // encoder sequence length (fixed by reference)
#define BT 64      // batch rows per block
#define NTHREADS 128

typedef __attribute__((ext_vector_type(16))) _Float16 v16h;
typedef __attribute__((ext_vector_type(8)))  float    v8f;

__device__ __forceinline__ v8f wmma16(v16h a, v16h b, v8f c) {
  return __builtin_amdgcn_wmma_f32_16x16x32_f16(false, a, false, b, (short)0, c,
                                                false, false);
}

// B-fragment (32x16 K-major per lane-half): lane = {hl=lane/16, n=lane%16},
// element e -> K = 32*kf + (e<8 ? e : e+8) + 8*hl. Stored pre-swizzled in LDS: 16 halves
// contiguous per lane -> one aligned 32B vector load.
__device__ __forceinline__ v16h ldB(const _Float16* w, int frag, int lane) {
  return *(const v16h*)(w + frag * 512 + lane * 16);
}

// A-fragment (16x32 f16, ISA 7.12.2): lane row m = lane%16, hl = lane/16;
// element e -> K = kbase + (e<8 ? e : e+8) + 8*hl.
__device__ __forceinline__ v16h mkA(const float* row, int kbase, int hl) {
  v16h a;
#pragma unroll
  for (int e = 0; e < 16; ++e) {
    int k = kbase + ((e < 8) ? e : (e + 8)) + 8 * hl;
    a[e] = (_Float16)row[k];
  }
  return a;
}

__device__ __forceinline__ float sigmf(float v) { return 1.f / (1.f + __expf(-v)); }
__device__ __forceinline__ float tanh_fast(float v) {
  float e = __expf(2.f * v);
  return (e - 1.f) / (e + 1.f);
}

// Stage a weight matrix W[n, k] (row-major, n<192, k<K) into LDS as
// pre-swizzled f16 B-fragments for out = in @ W^T.
__device__ void fillB(_Float16* dst, const float* __restrict__ W, int K,
                      int kfrags, int tid) {
  int total = NTILES * kfrags * 512;
  for (int i = tid; i < total; i += NTHREADS) {
    int e    = i & 15;
    int lane = (i >> 4) & 31;
    int frag = i >> 9;
    int kf = frag % kfrags;
    int nt = frag / kfrags;
    int k = kf * 32 + ((e < 8) ? e : (e + 8)) + 8 * (lane >> 4);
    int n = nt * 16 + (lane & 15);
    float v = (k < K) ? W[n * K + k] : 0.f;
    dst[i] = (_Float16)v;
  }
}

// ---------------- Encoder: persistent per-block time loop ----------------
__global__ __launch_bounds__(NTHREADS) void enc_kernel(
    const float* __restrict__ x,
    const float* __restrict__ Wih0, const float* __restrict__ Whh0,
    const float* __restrict__ bih0, const float* __restrict__ bhh0,
    const float* __restrict__ Wih1, const float* __restrict__ Whh1,
    const float* __restrict__ bih1, const float* __restrict__ bhh1,
    float* __restrict__ hout, int B, int NI) {
  extern __shared__ __align__(32) char smem[];
  _Float16* sWih0 = (_Float16*)smem;            // 12 frags (K=NI padded to 32)
  _Float16* sWhh0 = sWih0 + 12 * 512;           // 24 frags (K=64)
  _Float16* sWih1 = sWhh0 + 24 * 512;           // 24 frags
  _Float16* sWhh1 = sWih1 + 24 * 512;           // 24 frags
  float* brz0 = (float*)(sWhh1 + 24 * 512);     // 128: bih+bhh for r,z cols
  float* bin0 = brz0 + 128;                     // 64: bih for n cols
  float* bhn0 = bin0 + 64;                      // 64: bhh for n cols
  float* brz1 = bhn0 + 64;
  float* bin1 = brz1 + 128;
  float* bhn1 = bin1 + 64;
  float* sH0 = bhn1 + 64;                       // [BT][64] f32
  float* sH1 = sH0 + BT * HD;

  const int tid = threadIdx.x;
  fillB(sWih0, Wih0, NI, 1, tid);
  fillB(sWhh0, Whh0, HD, 2, tid);
  fillB(sWih1, Wih1, HD, 2, tid);
  fillB(sWhh1, Whh1, HD, 2, tid);
  for (int i = tid; i < 128; i += NTHREADS) {
    brz0[i] = bih0[i] + bhh0[i];
    brz1[i] = bih1[i] + bhh1[i];
  }
  for (int i = tid; i < 64; i += NTHREADS) {
    bin0[i] = bih0[128 + i]; bhn0[i] = bhh0[128 + i];
    bin1[i] = bih1[128 + i]; bhn1[i] = bhh1[128 + i];
  }
  for (int i = tid; i < BT * HD; i += NTHREADS) { sH0[i] = 0.f; sH1[i] = 0.f; }
  __syncthreads();

  const int wv = tid >> 5, lane = tid & 31;
  const int m = lane & 15, hl = lane >> 4;
  float* h0w = sH0 + wv * 16 * HD;   // this wave's 16 rows, no cross-wave deps
  float* h1w = sH1 + wv * 16 * HD;
  const long brow = (long)blockIdx.x * BT + wv * 16 + m;
  const float* xr = x + brow * (long)TSEQ * NI;

  for (int t = 0; t < TSEQ; ++t) {
    // ---- layer 0 ----
    v16h ax;  // x_t as 16x32 A-frag, K>=NI zero-padded
#pragma unroll
    for (int e = 0; e < 16; ++e) {
      int k = ((e < 8) ? e : (e + 8)) + 8 * hl;
      ax[e] = (k < NI) ? (_Float16)xr[t * NI + k] : (_Float16)0.f;
    }
    if (t + 1 < TSEQ) __builtin_prefetch(xr + (t + 1) * NI, 0, 1);
    v16h a0 = mkA(h0w + m * HD, 0, hl);
    v16h a1 = mkA(h0w + m * HD, 32, hl);
    v8f rz[8], gin[4], ghn[4];
#pragma unroll
    for (int nt = 0; nt < 8; ++nt) {  // r,z gates: merge x-part + h-part
      float bb = brz0[nt * 16 + m];
      v8f c;
#pragma unroll
      for (int e = 0; e < 8; ++e) c[e] = bb;
      c = wmma16(ax, ldB(sWih0, nt, lane), c);
      c = wmma16(a0, ldB(sWhh0, nt * 2 + 0, lane), c);
      c = wmma16(a1, ldB(sWhh0, nt * 2 + 1, lane), c);
      rz[nt] = c;
    }
#pragma unroll
    for (int j = 0; j < 4; ++j) {  // n gate: keep in / hn separate
      int nt = 8 + j;
      float bi = bin0[j * 16 + m], bh = bhn0[j * 16 + m];
      v8f ci, ch;
#pragma unroll
      for (int e = 0; e < 8; ++e) { ci[e] = bi; ch[e] = bh; }
      ci = wmma16(ax, ldB(sWih0, nt, lane), ci);
      ch = wmma16(a0, ldB(sWhh0, nt * 2 + 0, lane), ch);
      ch = wmma16(a1, ldB(sWhh0, nt * 2 + 1, lane), ch);
      gin[j] = ci; ghn[j] = ch;
    }
#pragma unroll
    for (int j = 0; j < 4; ++j) {  // h0 update (D layout: lane=col, e=row)
#pragma unroll
      for (int e = 0; e < 8; ++e) {
        int row = e + 8 * hl, col = j * 16 + m;
        float r = sigmf(rz[j][e]);
        float z = sigmf(rz[4 + j][e]);
        float nn = tanh_fast(gin[j][e] + r * ghn[j][e]);
        float hp = h0w[row * HD + col];
        h0w[row * HD + col] = (1.f - z) * nn + z * hp;
      }
    }
    // ---- layer 1 ----
    v16h c0 = mkA(h0w + m * HD, 0, hl);
    v16h c1 = mkA(h0w + m * HD, 32, hl);
    v16h d0 = mkA(h1w + m * HD, 0, hl);
    v16h d1 = mkA(h1w + m * HD, 32, hl);
#pragma unroll
    for (int nt = 0; nt < 8; ++nt) {
      float bb = brz1[nt * 16 + m];
      v8f c;
#pragma unroll
      for (int e = 0; e < 8; ++e) c[e] = bb;
      c = wmma16(c0, ldB(sWih1, nt * 2 + 0, lane), c);
      c = wmma16(c1, ldB(sWih1, nt * 2 + 1, lane), c);
      c = wmma16(d0, ldB(sWhh1, nt * 2 + 0, lane), c);
      c = wmma16(d1, ldB(sWhh1, nt * 2 + 1, lane), c);
      rz[nt] = c;
    }
#pragma unroll
    for (int j = 0; j < 4; ++j) {
      int nt = 8 + j;
      float bi = bin1[j * 16 + m], bh = bhn1[j * 16 + m];
      v8f ci, ch;
#pragma unroll
      for (int e = 0; e < 8; ++e) { ci[e] = bi; ch[e] = bh; }
      ci = wmma16(c0, ldB(sWih1, nt * 2 + 0, lane), ci);
      ci = wmma16(c1, ldB(sWih1, nt * 2 + 1, lane), ci);
      ch = wmma16(d0, ldB(sWhh1, nt * 2 + 0, lane), ch);
      ch = wmma16(d1, ldB(sWhh1, nt * 2 + 1, lane), ch);
      gin[j] = ci; ghn[j] = ch;
    }
#pragma unroll
    for (int j = 0; j < 4; ++j) {
#pragma unroll
      for (int e = 0; e < 8; ++e) {
        int row = e + 8 * hl, col = j * 16 + m;
        float r = sigmf(rz[j][e]);
        float z = sigmf(rz[4 + j][e]);
        float nn = tanh_fast(gin[j][e] + r * ghn[j][e]);
        float hp = h1w[row * HD + col];
        h1w[row * HD + col] = (1.f - z) * nn + z * hp;
      }
    }
  }
  __syncthreads();
  // final hidden -> workspace for the decoder
  for (int i = tid; i < BT * HD; i += NTHREADS) {
    long gb = (long)blockIdx.x * BT + (i >> 6);
    hout[gb * HD + (i & 63)] = sH0[i];
    hout[(long)B * HD + gb * HD + (i & 63)] = sH1[i];
  }
}

// ---------------- Decoder: autoregressive persistent loop ----------------
__global__ __launch_bounds__(NTHREADS) void dec_kernel(
    const float* __restrict__ hin,
    const float* __restrict__ Wih0, const float* __restrict__ Whh0,
    const float* __restrict__ bih0, const float* __restrict__ bhh0,
    const float* __restrict__ Wih1, const float* __restrict__ Whh1,
    const float* __restrict__ bih1, const float* __restrict__ bhh1,
    const float* __restrict__ Won, const float* __restrict__ bon,
    const float* __restrict__ Wcv, const float* __restrict__ bcv,
    float* __restrict__ out, int B, int TLEN) {
  extern __shared__ __align__(32) char smem[];
  _Float16* sWhh0 = (_Float16*)smem;       // 24 frags
  _Float16* sWih1 = sWhh0 + 24 * 512;      // 24 frags
  _Float16* sWhh1 = sWih1 + 24 * 512;      // 24 frags
  float* w0   = (float*)(sWhh1 + 24 * 512);  // 192: Wih0 column (input size 1)
  float* brz0 = w0 + 192;
  float* bin0 = brz0 + 128;
  float* bhn0 = bin0 + 64;
  float* brz1 = bhn0 + 64;
  float* bin1 = brz1 + 128;
  float* bhn1 = bin1 + 64;
  float* sOn = bhn1 + 64;   // 64
  float* sCv = sOn + 64;    // 64
  float* sB2 = sCv + 64;    // {b_on, b_cv}
  float* sH0 = sB2 + 2;
  float* sH1 = sH0 + BT * HD;
  float* sPrev = sH1 + BT * HD;  // BT

  const int tid = threadIdx.x;
  fillB(sWhh0, Whh0, HD, 2, tid);
  fillB(sWih1, Wih1, HD, 2, tid);
  fillB(sWhh1, Whh1, HD, 2, tid);
  for (int i = tid; i < 192; i += NTHREADS) w0[i] = Wih0[i];
  for (int i = tid; i < 128; i += NTHREADS) {
    brz0[i] = bih0[i] + bhh0[i];
    brz1[i] = bih1[i] + bhh1[i];
  }
  for (int i = tid; i < 64; i += NTHREADS) {
    bin0[i] = bih0[128 + i]; bhn0[i] = bhh0[128 + i];
    bin1[i] = bih1[128 + i]; bhn1[i] = bhh1[128 + i];
    sOn[i] = Won[i]; sCv[i] = Wcv[i];
  }
  if (tid == 0) { sB2[0] = bon[0]; sB2[1] = bcv[0]; }
  for (int i = tid; i < BT * HD; i += NTHREADS) {
    long gb = (long)blockIdx.x * BT + (i >> 6);
    sH0[i] = hin[gb * HD + (i & 63)];
    sH1[i] = hin[(long)B * HD + gb * HD + (i & 63)];
  }
  for (int i = tid; i < BT; i += NTHREADS) sPrev[i] = 0.f;
  __syncthreads();

  const int wv = tid >> 5, lane = tid & 31;
  const int m = lane & 15, hl = lane >> 4;
  float* h0w = sH0 + wv * 16 * HD;
  float* h1w = sH1 + wv * 16 * HD;
  float* pw = sPrev + wv * 16;
  const long brow = (long)blockIdx.x * BT + wv * 16 + m;

  for (int t = 0; t < TLEN; ++t) {
    v16h a0 = mkA(h0w + m * HD, 0, hl);
    v16h a1 = mkA(h0w + m * HD, 32, hl);
    float pv[8];
#pragma unroll
    for (int e = 0; e < 8; ++e) pv[e] = pw[e + 8 * hl];
    v8f rz[8], gin[4], ghn[4];
#pragma unroll
    for (int nt = 0; nt < 8; ++nt) {  // layer 0: prev*w folded into C init
      int n = nt * 16 + m;
      float wn = w0[n], bb = brz0[n];
      v8f c;
#pragma unroll
      for (int e = 0; e < 8; ++e) c[e] = pv[e] * wn + bb;
      c = wmma16(a0, ldB(sWhh0, nt * 2 + 0, lane), c);
      c = wmma16(a1, ldB(sWhh0, nt * 2 + 1, lane), c);
      rz[nt] = c;
    }
#pragma unroll
    for (int j = 0; j < 4; ++j) {
      int nt = 8 + j;
      float wn = w0[nt * 16 + m];
      float bi = bin0[j * 16 + m], bh = bhn0[j * 16 + m];
      v8f ci, ch;
#pragma unroll
      for (int e = 0; e < 8; ++e) { ci[e] = pv[e] * wn + bi; ch[e] = bh; }
      ch = wmma16(a0, ldB(sWhh0, nt * 2 + 0, lane), ch);
      ch = wmma16(a1, ldB(sWhh0, nt * 2 + 1, lane), ch);
      gin[j] = ci; ghn[j] = ch;
    }
#pragma unroll
    for (int j = 0; j < 4; ++j) {
#pragma unroll
      for (int e = 0; e < 8; ++e) {
        int row = e + 8 * hl, col = j * 16 + m;
        float r = sigmf(rz[j][e]);
        float z = sigmf(rz[4 + j][e]);
        float nn = tanh_fast(gin[j][e] + r * ghn[j][e]);
        float hp = h0w[row * HD + col];
        h0w[row * HD + col] = (1.f - z) * nn + z * hp;
      }
    }
    // layer 1
    v16h c0 = mkA(h0w + m * HD, 0, hl);
    v16h c1 = mkA(h0w + m * HD, 32, hl);
    v16h d0 = mkA(h1w + m * HD, 0, hl);
    v16h d1 = mkA(h1w + m * HD, 32, hl);
#pragma unroll
    for (int nt = 0; nt < 8; ++nt) {
      float bb = brz1[nt * 16 + m];
      v8f c;
#pragma unroll
      for (int e = 0; e < 8; ++e) c[e] = bb;
      c = wmma16(c0, ldB(sWih1, nt * 2 + 0, lane), c);
      c = wmma16(c1, ldB(sWih1, nt * 2 + 1, lane), c);
      c = wmma16(d0, ldB(sWhh1, nt * 2 + 0, lane), c);
      c = wmma16(d1, ldB(sWhh1, nt * 2 + 1, lane), c);
      rz[nt] = c;
    }
#pragma unroll
    for (int j = 0; j < 4; ++j) {
      int nt = 8 + j;
      float bi = bin1[j * 16 + m], bh = bhn1[j * 16 + m];
      v8f ci, ch;
#pragma unroll
      for (int e = 0; e < 8; ++e) { ci[e] = bi; ch[e] = bh; }
      ci = wmma16(c0, ldB(sWih1, nt * 2 + 0, lane), ci);
      ci = wmma16(c1, ldB(sWih1, nt * 2 + 1, lane), ci);
      ch = wmma16(d0, ldB(sWhh1, nt * 2 + 0, lane), ch);
      ch = wmma16(d1, ldB(sWhh1, nt * 2 + 1, lane), ch);
      gin[j] = ci; ghn[j] = ch;
    }
#pragma unroll
    for (int j = 0; j < 4; ++j) {
#pragma unroll
      for (int e = 0; e < 8; ++e) {
        int row = e + 8 * hl, col = j * 16 + m;
        float r = sigmf(rz[j][e]);
        float z = sigmf(rz[4 + j][e]);
        float nn = tanh_fast(gin[j][e] + r * ghn[j][e]);
        float hp = h1w[row * HD + col];
        h1w[row * HD + col] = (1.f - z) * nn + z * hp;
      }
    }
    // heads: cv / on-logit dot over h1 row; gate sigmoid(lg)>0.5 <=> lg>0
    if (lane < 16) {
      const float* hr = h1w + m * HD;
      float lg = sB2[0], cv = sB2[1];
#pragma unroll 8
      for (int k = 0; k < HD; ++k) {
        float hv = hr[k];
        lg += hv * sOn[k];
        cv += hv * sCv[k];
      }
      float g = (lg > 0.f) ? cv : 0.f;
      pw[m] = g;
      out[brow * TLEN + t] = g;
    }
  }
}

extern "C" void kernel_launch(void* const* d_in, const int* in_sizes, int n_in,
                              void* d_out, int out_size, void* d_ws,
                              size_t ws_size, hipStream_t stream) {
  (void)n_in; (void)ws_size;
  const float* x = (const float*)d_in[0];
  int NI = in_sizes[2] / G3;                    // e_Wih0 is (3H, NI)
  long B = (long)in_sizes[0] / ((long)TSEQ * NI);
  int TLEN = (int)(out_size / B);               // target_len (deterministic)
  float* hbuf = (float*)d_ws;                   // [2, B, 64] f32 handoff

  size_t enc_sm = (size_t)(12 + 24 + 24 + 24) * 512 * 2 +
                  (size_t)(128 + 64 + 64) * 2 * 4 + (size_t)2 * BT * HD * 4;
  size_t dec_sm = (size_t)(24 * 3) * 512 * 2 +
                  (size_t)(192 + (128 + 64 + 64) * 2 + 64 + 64 + 2 + BT) * 4 +
                  (size_t)2 * BT * HD * 4;

  dim3 blk(NTHREADS);
  dim3 grd((unsigned)(B / BT));
  enc_kernel<<<grd, blk, enc_sm, stream>>>(
      x, (const float*)d_in[2], (const float*)d_in[3], (const float*)d_in[4],
      (const float*)d_in[5], (const float*)d_in[6], (const float*)d_in[7],
      (const float*)d_in[8], (const float*)d_in[9], hbuf, (int)B, NI);
  dec_kernel<<<grd, blk, dec_sm, stream>>>(
      hbuf, (const float*)d_in[10], (const float*)d_in[11],
      (const float*)d_in[12], (const float*)d_in[13], (const float*)d_in[14],
      (const float*)d_in[15], (const float*)d_in[16], (const float*)d_in[17],
      (const float*)d_in[18], (const float*)d_in[19], (const float*)d_in[20],
      (const float*)d_in[21], (float*)d_out, (int)B, TLEN);
}
